// DIMKT_37323265802964
// MI455X (gfx1250) — matrix operand order repaired and checked by reading, not compiled, not run on hardware
//
#include <hip/hip_runtime.h>
#include <hip/hip_bf16.h>

typedef _Float16 v16h __attribute__((ext_vector_type(16)));
typedef _Float16 v8h  __attribute__((ext_vector_type(8)));
typedef float    v8f  __attribute__((ext_vector_type(8)));

#define BB 256
#define SS 512
#define DD 128
#define ROWS_PER_BLK 16
#define ASTRIDE 136   // padded f16 row stride (halves) for A-operand buffers -> no LDS bank conflicts

#define WMMA_F16(A,Bm,C) __builtin_amdgcn_wmma_f32_16x16x32_f16(false,(A),false,(Bm),(short)0,(C),false,false)

// branchless fast transcendentals: v_exp_f32 + v_rcp_f32, no EXEC-divergent paths
__device__ __forceinline__ float sigmoidf_(float x){
  return __builtin_amdgcn_rcpf(1.0f + __expf(-x));
}
__device__ __forceinline__ float tanhf_(float x){
  // 1 - 2/(e^{2x}+1): x->+inf => 1, x->-inf => -1, monotone, branch-free
  return 1.0f - 2.0f * __builtin_amdgcn_rcpf(__expf(2.0f * x) + 1.0f);
}

// ---------------- dynamic LDS layout (bytes) ----------------
// packed f16 B-operand weights: W_s1, W_s2, W_p1[:D], W_p2[:D], W_ki[:D]
constexpr int OFF_WPACK = 0;                         // 5 * 16384 halves = 163840
constexpr int OFF_H     = 163840;                    // h fp32 16x128      8192
constexpr int OFF_X0    = 172032;                    // x buf0 fp32        8192
constexpr int OFF_X1    = 180224;                    // x buf1 fp32        8192
constexpr int OFF_D     = 188416;                    // d f16 16x136       4352
constexpr int OFF_SDF   = 192768;                    // sdf f16 16x136     4352
constexpr int OFF_HF    = 197120;                    // h f16 16x136       4352
constexpr int OFF_PC1   = 201472;                    // 2x128 fp32         1024
constexpr int OFF_PC2   = 202496;                    // 2x128 fp32         1024
constexpr int OFF_KC    = 203520;                    // 2x128 fp32         1024
constexpr int OFF_BS1   = 204544;                    // 128 fp32            512
constexpr int OFF_BS2   = 205056;                    // 128 fp32            512
constexpr int OFF_IDX   = 205568;                    // 3x16 int            256
constexpr int OFF_KQD   = 205824;                    // 101x128 fp32      51712
constexpr int OFF_KCD   = 257536;                    // 101x128 fp32      51712
constexpr int SMEM_BYTES= 309248;                    // < 320KB per-WG limit

// ---- WMMA fragment loaders (16x16x32 f16) ----
// B-matrix (32x16 K-major tile) packed layout: halves j=0..15 of lane L hold
// W[kt*32 + (L>>4)*16 + j][ntile*16 + (L&15)]
__device__ __forceinline__ v16h loadB(const _Float16* wbase, int ntile, int kt, int lane){
  const _Float16* p = wbase + ((((ntile<<2) + kt)<<5) + lane)*16;
  v8h lo = *(const v8h*)p;
  v8h hi = *(const v8h*)(p + 8);
  return __builtin_shufflevector(lo, hi, 0,1,2,3,4,5,6,7,8,9,10,11,12,13,14,15);
}
// A-matrix (16x32 tile) from row-major f16 buffer with ASTRIDE row pitch:
// lane L (M = L&15, half = L>>4): halves j=0..15 -> K = kt*32 + half*16 + j
__device__ __forceinline__ v16h loadA(const _Float16* abuf, int kt, int lane){
  const int m = lane & 15, half = lane >> 4;
  const _Float16* p = abuf + m*ASTRIDE + kt*32 + half*16;
  v8h lo = *(const v8h*)p;
  v8h hi = *(const v8h*)(p + 8);
  return __builtin_shufflevector(lo, hi, 0,1,2,3,4,5,6,7,8,9,10,11,12,13,14,15);
}

// x_t[r, :] = Xq[q[r,t]] + Xc[c[r,t]] + Xqd[qd[r,t]] + Xcd[cd[r,t]]   (bx folded into Xcd)
__device__ __forceinline__ void gather_x(float* dst, int tt, int rbase, int tid,
    const int* __restrict__ qseq, const int* __restrict__ cseq,
    const int* __restrict__ qdseq, const int* __restrict__ cdseq,
    const float* __restrict__ Xq, const float* __restrict__ Xc,
    const float* __restrict__ Xqd, const float* __restrict__ Xcd)
{
  const int r = tid >> 4, c0 = (tid & 15) * 8;
  const int g = (rbase + r) * SS + tt;
  const int qi = qseq[g], ci = cseq[g], di = qdseq[g], ei = cdseq[g];
  const float4* pq = (const float4*)(Xq  + qi*DD + c0);
  const float4* pc = (const float4*)(Xc  + ci*DD + c0);
  const float4* pd = (const float4*)(Xqd + di*DD + c0);
  const float4* pe = (const float4*)(Xcd + ei*DD + c0);
  float4* po = (float4*)(dst + r*DD + c0);
#pragma unroll
  for (int j = 0; j < 2; ++j){
    float4 a = pq[j], b = pc[j], c = pd[j], d = pe[j];
    float4 s;
    s.x = a.x + b.x + c.x + d.x;  s.y = a.y + b.y + c.y + d.y;
    s.z = a.z + b.z + c.z + d.z;  s.w = a.w + b.w + c.w + d.w;
    po[j] = s;
  }
}

// ---------------- precompute: T[i,:] = E[i,:] @ W[kOff:kOff+128, :] (+ bias) ----------------
__global__ __launch_bounds__(128)
void tab_gemm(const float* __restrict__ E, const float* __restrict__ W,
              const float* __restrict__ bias, float* __restrict__ T, int kOff)
{
  __shared__ float er[DD];
  const int i = blockIdx.x, n = threadIdx.x;
  er[n] = E[i*DD + n];
  __syncthreads();
  float acc = bias ? bias[n] : 0.0f;
#pragma unroll 8
  for (int k = 0; k < DD; ++k)
    acc = fmaf(er[k], W[(kOff + k)*DD + n], acc);
  T[i*DD + n] = acc;
}

// ---------------- persistent recurrence kernel: 16 blocks x 16 rows x 511 steps ----------------
__global__ __launch_bounds__(256, 1)
void dimkt_rnn(const int* __restrict__ qseq, const int* __restrict__ cseq,
               const int* __restrict__ qdseq, const int* __restrict__ cdseq,
               const int* __restrict__ corrseq,
               const float* __restrict__ Xq, const float* __restrict__ Xc,
               const float* __restrict__ Xqd, const float* __restrict__ Xcd,
               const float* __restrict__ Pc1g, const float* __restrict__ Pc2g,
               const float* __restrict__ Kcg,  const float* __restrict__ Kqdg,
               const float* __restrict__ Kcdg,
               const float* __restrict__ W_s1, const float* __restrict__ b_s1,
               const float* __restrict__ W_s2, const float* __restrict__ b_s2,
               const float* __restrict__ W_p1, const float* __restrict__ W_p2,
               const float* __restrict__ W_ki,
               const float* __restrict__ h0,
               float* __restrict__ y)
{
  extern __shared__ char smem[];
  _Float16* wp   = (_Float16*)(smem + OFF_WPACK);
  float* hbuf    = (float*)(smem + OFF_H);
  float* xb0     = (float*)(smem + OFF_X0);
  float* xb1     = (float*)(smem + OFF_X1);
  _Float16* dbuf = (_Float16*)(smem + OFF_D);
  _Float16* sdfb = (_Float16*)(smem + OFF_SDF);
  _Float16* hf16 = (_Float16*)(smem + OFF_HF);
  float* Pc1 = (float*)(smem + OFF_PC1);
  float* Pc2 = (float*)(smem + OFF_PC2);
  float* Kc  = (float*)(smem + OFF_KC);
  float* bs1 = (float*)(smem + OFF_BS1);
  float* bs2 = (float*)(smem + OFF_BS2);
  int* idxC  = (int*)(smem + OFF_IDX);
  int* idxQ  = idxC + 16;
  int* idxD  = idxC + 32;
  float* Kqd = (float*)(smem + OFF_KQD);
  float* Kcd = (float*)(smem + OFF_KCD);

  const int tid   = threadIdx.x;
  const int rbase = blockIdx.x * ROWS_PER_BLK;

  // ---- one-time: pack the five 128x128 weight matrices into f16 B-fragment layout ----
  const float* srcs[5] = { W_s1, W_s2, W_p1, W_p2, W_ki };
#pragma unroll
  for (int m = 0; m < 5; ++m){
    const float* Wsrc = srcs[m];
    _Float16* dst = wp + m*16384;
    for (int e = tid; e < 16384; e += 256){
      const int j    = e & 15;
      const int lane = (e >> 4) & 31;
      const int kt   = (e >> 9) & 3;
      const int nt   = e >> 11;
      const int k = kt*32 + (lane >> 4)*16 + j;
      const int n = nt*16 + (lane & 15);
      dst[e] = (_Float16)Wsrc[k*DD + n];
    }
  }
  // gate lookup tables into LDS (removes all per-step global latency from the update phase)
  for (int e = tid; e < 101*DD; e += 256){ Kqd[e] = Kqdg[e]; Kcd[e] = Kcdg[e]; }
  // small tables + biases into LDS
  Pc1[tid] = Pc1g[tid];  Pc2[tid] = Pc2g[tid];  Kc[tid] = Kcg[tid];
  if (tid < DD){ bs1[tid] = b_s1[tid]; bs2[tid] = b_s2[tid]; }
  // h0 init (fp32 + f16 copy), zero last y column
  {
    const int r = tid >> 4, c0 = (tid & 15) * 8;
#pragma unroll
    for (int j = 0; j < 8; ++j){
      float v = h0[(rbase + r)*DD + c0 + j];
      hbuf[r*DD + c0 + j]      = v;
      hf16[r*ASTRIDE + c0 + j] = (_Float16)v;
    }
    if (tid < ROWS_PER_BLK) y[(rbase + tid)*SS + (SS - 1)] = 0.0f;
  }
  gather_x(xb0, 0, rbase, tid, qseq, cseq, qdseq, cdseq, Xq, Xc, Xqd, Xcd);
  gather_x(xb1, 1, rbase, tid, qseq, cseq, qdseq, cdseq, Xq, Xc, Xqd, Xcd);
  __syncthreads();

  const int w    = tid >> 5;          // wave -> N tile (cols 16w..16w+15)
  const int lane = tid & 31;
  const int col  = w*16 + (lane & 15);
  const int rb   = (lane >> 4) * 8;   // C-tile row base for this lane
  const float mybs1 = bs1[col];
  const float mybs2 = bs2[col];

  for (int t = 0; t < SS - 1; ++t){
    float* xc = (t & 1) ? xb1 : xb0;
    float* xn = (t & 1) ? xb0 : xb1;

    // d = x_t - h  (f16 to LDS); load step indices; prefetch x_{t+2} into xc slot
    {
      const int r = tid >> 4, c0 = (tid & 15) * 8;
#pragma unroll
      for (int j = 0; j < 8; ++j)
        dbuf[r*ASTRIDE + c0 + j] = (_Float16)(xc[r*DD + c0 + j] - hbuf[r*DD + c0 + j]);
      if (tid < ROWS_PER_BLK){
        const int g = (rbase + tid)*SS + t;
        idxC[tid] = corrseq[g];  idxQ[tid] = qdseq[g];  idxD[tid] = cdseq[g];
      }
      if (t + 2 < SS)
        gather_x(xc, t + 2, rbase, tid, qseq, cseq, qdseq, cdseq, Xq, Xc, Xqd, Xcd);
    }
    __syncthreads();

    // stage 1: s1 = d@W_s1, s2 = d@W_s2, g_pre = h@W_ki_h   (K=128, 4 wmma each)
    v8f acc1 = {}; v8f acc2 = {}; v8f accg = {};
#pragma unroll
    for (int kt = 0; kt < 4; ++kt){
      v16h aD = loadA(dbuf, kt, lane);
      v16h aH = loadA(hf16, kt, lane);
      v16h b1 = loadB(wp +      0, w, kt, lane);
      v16h b2 = loadB(wp +  16384, w, kt, lane);
      v16h bk = loadB(wp +  65536, w, kt, lane);
      acc1 = WMMA_F16(aD, b1, acc1);
      acc2 = WMMA_F16(aD, b2, acc2);
      accg = WMMA_F16(aH, bk, accg);
    }
    // sdf = sigmoid(s1+b1)*tanh(s2+b2), written in C-tile layout
#pragma unroll
    for (int v = 0; v < 8; ++v){
      float sv = sigmoidf_(acc1[v] + mybs1) * tanhf_(acc2[v] + mybs2);
      sdfb[(rb + v)*ASTRIDE + col] = (_Float16)sv;
    }
    __syncthreads();

    // stage 2: p1 = sdf@W_p1a, p2 = sdf@W_p2a
    v8f acc3 = {}; v8f acc4 = {};
#pragma unroll
    for (int kt = 0; kt < 4; ++kt){
      v16h aS = loadA(sdfb, kt, lane);
      v16h b3 = loadB(wp + 32768, w, kt, lane);
      v16h b4 = loadB(wp + 49152, w, kt, lane);
      acc3 = WMMA_F16(aS, b3, acc3);
      acc4 = WMMA_F16(aS, b4, acc4);
    }
    // gate + state update (each (row,col) element owned by exactly one lane; all LDS)
#pragma unroll
    for (int v = 0; v < 8; ++v){
      const int r  = rb + v;
      const int ci = idxC[r];
      float pka = sigmoidf_(acc3[v] + Pc1[ci*DD + col]) * tanhf_(acc4[v] + Pc2[ci*DD + col]);
      float g   = sigmoidf_(accg[v] + Kc[ci*DD + col]
                            + Kqd[idxQ[r]*DD + col] + Kcd[idxD[r]*DD + col]);
      float ho = hbuf[r*DD + col];
      float hn = g*ho + (1.0f - g)*pka;
      hbuf[r*DD + col]      = hn;
      hf16[r*ASTRIDE + col] = (_Float16)hn;
    }
    __syncthreads();

    // y[b,t] = sigmoid(dot(x_{t+1}, h_new))  -- 16-lane tree reduction per row
    {
      const int r = tid >> 4, c0 = (tid & 15) * 8;
      float p = 0.0f;
#pragma unroll
      for (int j = 0; j < 8; ++j)
        p += xn[r*DD + c0 + j] * hbuf[r*DD + c0 + j];
      p += __shfl_xor(p, 8, 32);
      p += __shfl_xor(p, 4, 32);
      p += __shfl_xor(p, 2, 32);
      p += __shfl_xor(p, 1, 32);
      if ((tid & 15) == 0) y[(rbase + r)*SS + t] = sigmoidf_(p);
    }
    // no barrier needed: every LDS region rewritten next iteration is either
    // thread-private here or protected by next iteration's barriers.
  }
}

extern "C" void kernel_launch(void* const* d_in, const int* in_sizes, int n_in,
                              void* d_out, int out_size, void* d_ws, size_t ws_size,
                              hipStream_t stream)
{
  const int*   qseq    = (const int*)  d_in[0];
  const int*   cseq    = (const int*)  d_in[1];
  const int*   qdseq   = (const int*)  d_in[2];
  const int*   cdseq   = (const int*)  d_in[3];
  const int*   corrseq = (const int*)  d_in[4];
  const float* E_q     = (const float*)d_in[5];
  const float* E_c     = (const float*)d_in[6];
  const float* E_qd    = (const float*)d_in[7];
  const float* E_cd    = (const float*)d_in[8];
  const float* E_corr  = (const float*)d_in[9];
  const float* Wx      = (const float*)d_in[10];
  const float* bx      = (const float*)d_in[11];
  const float* W_s1    = (const float*)d_in[12];
  const float* b_s1    = (const float*)d_in[13];
  const float* W_s2    = (const float*)d_in[14];
  const float* b_s2    = (const float*)d_in[15];
  const float* W_p1    = (const float*)d_in[16];
  const float* b_p1    = (const float*)d_in[17];
  const float* W_p2    = (const float*)d_in[18];
  const float* b_p2    = (const float*)d_in[19];
  const float* W_ki    = (const float*)d_in[20];
  const float* b_ki    = (const float*)d_in[21];
  const float* h0      = (const float*)d_in[22];
  float* y = (float*)d_out;

  // workspace carve-up (floats)
  float* ws  = (float*)d_ws;
  float* Xq  = ws;                     // 20000*128
  float* Xc  = Xq  + 20000*DD;         //  2000*128
  float* Xqd = Xc  +  2000*DD;         //   101*128
  float* Xcd = Xqd +   101*DD;         //   101*128 (+bx)
  float* Pc1 = Xcd +   101*DD;         //     2*128 (+b_p1)
  float* Pc2 = Pc1 +     2*DD;         //     2*128 (+b_p2)
  float* Kc  = Pc2 +     2*DD;         //     2*128
  float* Kqd = Kc  +     2*DD;         //   101*128
  float* Kcd = Kqd +   101*DD;         //   101*128 (+b_ki)

  // precompute lookup tables (replaces the big gather+GEMM and concat halves)
  tab_gemm<<<20000, 128, 0, stream>>>(E_q,    Wx,   nullptr, Xq,  0);
  tab_gemm<<< 2000, 128, 0, stream>>>(E_c,    Wx,   nullptr, Xc,  128);
  tab_gemm<<<  101, 128, 0, stream>>>(E_qd,   Wx,   nullptr, Xqd, 256);
  tab_gemm<<<  101, 128, 0, stream>>>(E_cd,   Wx,   bx,      Xcd, 384);
  tab_gemm<<<    2, 128, 0, stream>>>(E_corr, W_p1, b_p1,    Pc1, 128);
  tab_gemm<<<    2, 128, 0, stream>>>(E_corr, W_p2, b_p2,    Pc2, 128);
  tab_gemm<<<    2, 128, 0, stream>>>(E_corr, W_ki, nullptr, Kc,  128);
  tab_gemm<<<  101, 128, 0, stream>>>(E_qd,   W_ki, nullptr, Kqd, 256);
  tab_gemm<<<  101, 128, 0, stream>>>(E_cd,   W_ki, b_ki,    Kcd, 384);

  (void)hipFuncSetAttribute((const void*)dimkt_rnn,
                            hipFuncAttributeMaxDynamicSharedMemorySize, SMEM_BYTES);

  dimkt_rnn<<<BB / ROWS_PER_BLK, 256, SMEM_BYTES, stream>>>(
      qseq, cseq, qdseq, cdseq, corrseq,
      Xq, Xc, Xqd, Xcd, Pc1, Pc2, Kc, Kqd, Kcd,
      W_s1, b_s1, W_s2, b_s2, W_p1, W_p2, W_ki, h0, y);
}
